// TrafficGNN_31971736551689
// MI455X (gfx1250) — matrix operand order repaired and checked by reading, not compiled, not run on hardware
//
#include <hip/hip_runtime.h>
#include <hip/hip_bf16.h>
#include <hip/hip_fp16.h>

// ---------------------------------------------------------------------------
// TrafficGNN on MI455X (gfx1250): 2x SAGEConv + mean pool + linear head.
//  * GEMMs: v_wmma_f32_16x16x32_f16, A-tiles staged in LDS (8x reuse/block)
//  * Edge scatter: packed f16 atomics (global_atomic_pk_add_f16) -> half the
//    HBM traffic of the f32 version (scatter is the roofline bottleneck)
// ---------------------------------------------------------------------------

#define N_NODES  50000
#define IN_CH    64
#define HID      128
#define N_EDGES  800000
#define N_GRAPHS 64

typedef __attribute__((ext_vector_type(16))) _Float16 v16h;
typedef __attribute__((ext_vector_type(8)))  _Float16 v8h;
typedef __attribute__((ext_vector_type(8)))  float    v8f;

// ----------------------------- utility kernels -----------------------------

__global__ void zero_f32_kernel(float* __restrict__ p, int n) {
    int i = blockIdx.x * blockDim.x + threadIdx.x;
    if (i < n) p[i] = 0.0f;
}

__global__ void f32_to_f16_kernel(const float* __restrict__ src,
                                  _Float16* __restrict__ dst, int n) {
    int i = blockIdx.x * blockDim.x + threadIdx.x;
    if (i < n) dst[i] = (_Float16)src[i];
}

// W [K,H] fp32 row-major  ->  WT [H,K] f16 row-major (so each WMMA B-fragment
// per lane is one contiguous 32B run).
__global__ void transpose_to_f16_kernel(const float* __restrict__ W,
                                        _Float16* __restrict__ WT,
                                        int K, int H) {
    int i = blockIdx.x * blockDim.x + threadIdx.x;
    if (i < K * H) {
        int k = i / H, h = i % H;
        WT[h * K + k] = (_Float16)W[k * H + h];
    }
}

// Scatter-add f16 source features into f16 accumulators with packed atomics.
// One thread per (edge, channel-pair). cnt may be null (degree reused).
__global__ void scatter_pk_add_kernel(const __half2* __restrict__ feat,
                                      const int* __restrict__ src,
                                      const int* __restrict__ dst,
                                      __half2* __restrict__ aggr,
                                      float* __restrict__ cnt,
                                      int C2, long long total) {
    long long i = (long long)blockIdx.x * blockDim.x + threadIdx.x;
    if (i >= total) return;
    int e = (int)(i / C2);
    int c = (int)(i - (long long)e * C2);
    int s = src[e], d = dst[e];
    atomicAdd(&aggr[(long long)d * C2 + c], feat[(long long)s * C2 + c]);
    if (cnt && c == 0) atomicAdd(&cnt[d], 1.0f);
}

// In-place mean: aggr[i] /= max(cnt,1)
__global__ void mean_inplace_f16_kernel(_Float16* __restrict__ aggr,
                                        const float* __restrict__ cnt,
                                        int C, long long total) {
    long long i = (long long)blockIdx.x * blockDim.x + threadIdx.x;
    if (i >= total) return;
    int n = (int)(i / C);
    float d = cnt[n];
    d = d > 1.0f ? d : 1.0f;
    aggr[i] = (_Float16)((float)aggr[i] / d);
}

// ------------------------------ WMMA fragments ------------------------------

// A fragment in this code's contiguous k order (matches B): lane&15 = row M; lanes 0-15 hold
// K = kk+0..15, lanes 16-31 hold K = kk+16..31.
__device__ __forceinline__ v16h load_a_frag_lds(const _Float16* rowptr,
                                                int kk, int lane) {
    int kbase = (lane >> 4) << 4;              // 0 or 16
    v8h lo = *(const v8h*)(rowptr + kk + kbase);      // ds_load_b128
    v8h hi = *(const v8h*)(rowptr + kk + kbase + 8);  // ds_load_b128
    v16h a;
#pragma unroll
    for (int i = 0; i < 8; ++i) { a[i] = lo[i]; a[i + 8] = hi[i]; }
    return a;
}

// B fragment in this code's contiguous k order (matches A): lane&15 = col N; lanes 0-15 hold K=kk+0..15,
// lanes 16-31 hold K=kk+16..31. With WT in [H,K] this is contiguous.
__device__ __forceinline__ v16h load_b_frag(const _Float16* __restrict__ WT,
                                            int K, int col, int kk, int lane) {
    int kb = (lane >> 4) << 4;                 // 0 or 16
    return *(const v16h*)(WT + (long long)col * K + kk + kb);
}

// ------------------------------- fused GEMM --------------------------------
// out[N,128] = relu(A1[N,K] @ W1T' + A2[N,K] @ W2T' + bias).
// One wave per 16x16 tile; block = 256 thr = 8 waves = 8 column tiles of one
// 16-row stripe. All 8 waves share the same 16 A-rows -> stage A1/A2 tiles in
// LDS once per block (8x traffic reduction). N % 16 == 0 -> no guards, EXEC
// stays all-ones (WMMA requirement).
template <int K, bool WRITE_F32, bool WRITE_H16>
__global__ void sage_gemm_kernel(const _Float16* __restrict__ A1,   // [N,K] aggr
                                 const _Float16* __restrict__ A2,   // [N,K] self
                                 const _Float16* __restrict__ W1T,  // [128,K]
                                 const _Float16* __restrict__ W2T,  // [128,K]
                                 const float* __restrict__ bias,    // [128]
                                 float* __restrict__ outF,          // [N,128]
                                 _Float16* __restrict__ outH) {     // [N,128]
    __shared__ alignas(16) _Float16 lA1[16 * K];
    __shared__ alignas(16) _Float16 lA2[16 * K];

    const int tid  = threadIdx.x;
    const int lane = tid & 31;
    const int wave = tid >> 5;                 // 0..7 -> column tile
    const int m0   = blockIdx.x << 4;          // row tile base
    const int col  = (wave << 4) + (lane & 15);

    // Cooperative stage of the 16xK A tiles (b128 global loads -> b128 LDS).
    {
        const v8h* g1 = (const v8h*)(A1 + (long long)m0 * K);
        const v8h* g2 = (const v8h*)(A2 + (long long)m0 * K);
        constexpr int CHUNKS = (16 * K) / 8;   // v8h chunks per tile
#pragma unroll
        for (int c = tid; c < CHUNKS; c += 256) {
            ((v8h*)lA1)[c] = g1[c];
            ((v8h*)lA2)[c] = g2[c];
        }
    }
    __syncthreads();

    const _Float16* a1row = lA1 + (lane & 15) * K;
    const _Float16* a2row = lA2 + (lane & 15) * K;

    v8f acc = {};
#pragma unroll
    for (int kk = 0; kk < K; kk += 32) {
        __builtin_prefetch(W1T + (long long)col * K + kk + 32, 0, 0); // global_prefetch_b8
        __builtin_prefetch(W2T + (long long)col * K + kk + 32, 0, 0);
        v16h a1 = load_a_frag_lds(a1row, kk, lane);
        v16h b1 = load_b_frag(W1T, K, col, kk, lane);
        acc = __builtin_amdgcn_wmma_f32_16x16x32_f16(
                  false, a1, false, b1, (short)0, acc, false, false);
        v16h a2 = load_a_frag_lds(a2row, kk, lane);
        v16h b2 = load_b_frag(W2T, K, col, kk, lane);
        acc = __builtin_amdgcn_wmma_f32_16x16x32_f16(
                  false, a2, false, b2, (short)0, acc, false, false);
    }

    // C/D layout: element i -> row m0 + i + 8*(lane>>4), col = (wave<<4)+(lane&15)
    const float bv = bias[col];
    const int rbase = m0 + ((lane >> 4) << 3);
#pragma unroll
    for (int i = 0; i < 8; ++i) {
        float v = acc[i] + bv;
        v = v > 0.0f ? v : 0.0f;
        long long o = (long long)(rbase + i) * HID + col;
        if (WRITE_F32) outF[o] = v;
        if (WRITE_H16) outH[o] = (_Float16)v;
    }
}

// ------------------------------ pooling + head ------------------------------

__global__ void pool_add_kernel(const float* __restrict__ h,      // [N,128]
                                const int* __restrict__ batch,    // [N]
                                float* __restrict__ gs,           // [G,128]
                                float* __restrict__ gc) {         // [G]
    long long i = (long long)blockIdx.x * blockDim.x + threadIdx.x;
    if (i >= (long long)N_NODES * HID) return;
    int n = (int)(i >> 7);
    int c = (int)(i & (HID - 1));
    int g = batch[n];
    atomicAdd(&gs[g * HID + c], h[i]);
    if (c == 0) atomicAdd(&gc[g], 1.0f);
}

__global__ void head_kernel(const float* __restrict__ gs,
                            const float* __restrict__ gc,
                            const float* __restrict__ Wout,   // [128]
                            const float* __restrict__ bout,   // [1]
                            float* __restrict__ out) {        // [64]
    int g = threadIdx.x;
    if (g >= N_GRAPHS) return;
    float d = gc[g];
    d = d > 1.0f ? d : 1.0f;
    float acc = 0.0f;
#pragma unroll 8
    for (int k = 0; k < HID; ++k) acc += (gs[g * HID + k] / d) * Wout[k];
    out[g] = acc + bout[0];
}

// --------------------------------- launcher ---------------------------------

static inline size_t align256(size_t x) { return (x + 255) & ~(size_t)255; }
static inline int gblk(long long n) { return (int)((n + 255) / 256); }

extern "C" void kernel_launch(void* const* d_in, const int* in_sizes, int n_in,
                              void* d_out, int out_size, void* d_ws, size_t ws_size,
                              hipStream_t stream) {
    const float* x     = (const float*)d_in[0];
    const int*   edge  = (const int*)d_in[1];   // [2, E]
    const int*   srcI  = edge;
    const int*   dstI  = edge + N_EDGES;
    const int*   batch = (const int*)d_in[2];
    const float* Wl1   = (const float*)d_in[3];
    const float* bl1   = (const float*)d_in[4];
    const float* Wr1   = (const float*)d_in[5];
    const float* Wl2   = (const float*)d_in[6];
    const float* bl2   = (const float*)d_in[7];
    const float* Wr2   = (const float*)d_in[8];
    const float* Wout  = (const float*)d_in[9];
    const float* bout  = (const float*)d_in[10];
    float* out = (float*)d_out;

    // -------- workspace carve-up (~58 MB) --------
    char* ws = (char*)d_ws;
    _Float16* aggrH = (_Float16*)ws; ws += align256((size_t)N_NODES * HID * 2); // f16 accum, layer1 uses [N,64]
    float*    cntF  = (float*)ws;    ws += align256((size_t)N_NODES * 4);
    _Float16* xH    = (_Float16*)ws; ws += align256((size_t)N_NODES * IN_CH * 2);
    _Float16* h1H   = (_Float16*)ws; ws += align256((size_t)N_NODES * HID * 2);
    float*    h2F   = (float*)ws;    ws += align256((size_t)N_NODES * HID * 4);
    _Float16* wl1T  = (_Float16*)ws; ws += align256((size_t)IN_CH * HID * 2);
    _Float16* wr1T  = (_Float16*)ws; ws += align256((size_t)IN_CH * HID * 2);
    _Float16* wl2T  = (_Float16*)ws; ws += align256((size_t)HID * HID * 2);
    _Float16* wr2T  = (_Float16*)ws; ws += align256((size_t)HID * HID * 2);
    float*    gs    = (float*)ws;    ws += align256((size_t)N_GRAPHS * HID * 4);
    float*    gc    = (float*)ws;    ws += align256((size_t)N_GRAPHS * 4);

    const long long nh   = (long long)N_NODES * HID;     // 6.4M halfs (layer2 aggr)
    const long long nf   = (long long)N_NODES * IN_CH;   // 3.2M halfs (layer1 aggr)
    const long long e32  = (long long)N_EDGES * (IN_CH / 2);  // 25.6M pk-atomics
    const long long e64  = (long long)N_EDGES * (HID / 2);    // 51.2M pk-atomics

    // -------- init (zero f16 accumulator as f32 words) --------
    zero_f32_kernel<<<gblk(nh / 2), 256, 0, stream>>>((float*)aggrH, (int)(nh / 2));
    zero_f32_kernel<<<gblk(N_NODES), 256, 0, stream>>>(cntF, N_NODES);
    zero_f32_kernel<<<gblk(N_GRAPHS * HID), 256, 0, stream>>>(gs, N_GRAPHS * HID);
    zero_f32_kernel<<<1, 64, 0, stream>>>(gc, N_GRAPHS);

    // -------- precision conversion / weight transpose --------
    f32_to_f16_kernel<<<gblk(nf), 256, 0, stream>>>(x, xH, (int)nf);
    transpose_to_f16_kernel<<<gblk(IN_CH * HID), 256, 0, stream>>>(Wl1, wl1T, IN_CH, HID);
    transpose_to_f16_kernel<<<gblk(IN_CH * HID), 256, 0, stream>>>(Wr1, wr1T, IN_CH, HID);
    transpose_to_f16_kernel<<<gblk(HID * HID), 256, 0, stream>>>(Wl2, wl2T, HID, HID);
    transpose_to_f16_kernel<<<gblk(HID * HID), 256, 0, stream>>>(Wr2, wr2T, HID, HID);

    // -------- layer 1: f16 pk-atomic scatter-mean + fused WMMA GEMM + relu ----
    scatter_pk_add_kernel<<<gblk(e32), 256, 0, stream>>>(
        (const __half2*)xH, srcI, dstI, (__half2*)aggrH, cntF, IN_CH / 2, e32);
    mean_inplace_f16_kernel<<<gblk(nf), 256, 0, stream>>>(aggrH, cntF, IN_CH, nf);
    sage_gemm_kernel<IN_CH, false, true><<<N_NODES / 16, 256, 0, stream>>>(
        aggrH, xH, wl1T, wr1T, bl1, (float*)nullptr, h1H);

    // -------- layer 2 --------
    zero_f32_kernel<<<gblk(nh / 2), 256, 0, stream>>>((float*)aggrH, (int)(nh / 2));
    scatter_pk_add_kernel<<<gblk(e64), 256, 0, stream>>>(
        (const __half2*)h1H, srcI, dstI, (__half2*)aggrH, (float*)nullptr, HID / 2, e64);
    mean_inplace_f16_kernel<<<gblk(nh), 256, 0, stream>>>(aggrH, cntF, HID, nh);
    sage_gemm_kernel<HID, true, false><<<N_NODES / 16, 256, 0, stream>>>(
        aggrH, h1H, wl2T, wr2T, bl2, h2F, (_Float16*)nullptr);

    // -------- global mean pool + head --------
    pool_add_kernel<<<gblk(nh), 256, 0, stream>>>(h2F, batch, gs, gc);
    head_kernel<<<1, 64, 0, stream>>>(gs, gc, Wout, bout, out);
}
